// AKT_58918361366796
// MI455X (gfx1250) — compile-verified, matrix operand order checked by reading, not run on hardware
//
#include <hip/hip_runtime.h>

#define B_  16
#define S_  512
#define D_  512
#define H_  8
#define DK_ 64
#define FF_ 2048

typedef __attribute__((ext_vector_type(16))) _Float16 v16h;
typedef __attribute__((ext_vector_type(8)))  _Float16 v8h;
typedef __attribute__((ext_vector_type(8)))  float    v8f;
typedef __attribute__((ext_vector_type(4))) unsigned int u32x4;
typedef __attribute__((ext_vector_type(8))) int i32x8;
typedef __attribute__((ext_vector_type(4))) int i32x4;

static __device__ __forceinline__ v16h mk16(v8h lo, v8h hi) {
  v16h r;
#pragma unroll
  for (int i = 0; i < 8; i++) { r[i] = lo[i]; r[i + 8] = hi[i]; }
  return r;
}

static __device__ __forceinline__ float wave_max(float v) {
#pragma unroll
  for (int m = 16; m > 0; m >>= 1) v = fmaxf(v, __shfl_xor(v, m, 32));
  return v;
}
static __device__ __forceinline__ float wave_sum(float v) {
#pragma unroll
  for (int m = 16; m > 0; m >>= 1) v += __shfl_xor(v, m, 32);
  return v;
}
static __device__ __forceinline__ float wave_incl_scan(float v) {
  const int lane = threadIdx.x & 31;
#pragma unroll
  for (int d = 1; d < 32; d <<= 1) {
    float t = __shfl_up(v, d, 32);
    if (lane >= d) v += t;
  }
  return v;
}

// ---- Tensor Data Mover: 2-D f16 tile (tile_x cols, tile_y rows) global -> LDS.
// Optional LDS row padding: after every (pad_interval-coded) DWORDs insert
// (pad_amount-coded) DWORDs. D# layout per CDNA5 ISA ch.8.
static __device__ __forceinline__ void tdm_load_2d_f16(
    unsigned lds_off, const void* gptr,
    unsigned tile_x, unsigned tile_y,
    unsigned long long stride_elems,
    unsigned long long tensor_x, unsigned long long tensor_y,
    unsigned pad_interval_code, unsigned pad_amount_code, int pad_en) {
  unsigned long long ga = (unsigned long long)(size_t)gptr;
  union { unsigned long long q[2]; u32x4 v; } g0;
  g0.q[0] = 1ull /*count=1*/ | ((unsigned long long)lds_off << 32);
  g0.q[1] = (ga & ((1ull << 57) - 1)) | (2ull << 62); /*type=2 (image)*/
  union { unsigned long long q[4]; i32x8 v; } g1;
  unsigned w0 = (1u << 16) /*data_size=2B*/;
  if (pad_en) w0 |= (1u << 20) | (pad_interval_code << 22) | (pad_amount_code << 25);
  g1.q[0] = (unsigned long long)w0 | ((tensor_x & 0xFFFFull) << 48);
  g1.q[1] = ((tensor_x >> 16) & 0xFFFFull) | ((tensor_y & 0xFFFFFFFFull) << 16) |
            ((unsigned long long)(tile_x & 0xFFFF) << 48);
  g1.q[2] = (unsigned long long)(tile_y & 0xFFFF) | ((stride_elems & 0xFFFFFFFFull) << 32);
  g1.q[3] = (stride_elems >> 32) & 0xFFFFull;
  union { unsigned long long q[2]; i32x4 v; } g2;
  g2.q[0] = 1ull;  /* tensor_dim2 = 1 */
  g2.q[1] = 0ull;
  i32x4 g3 = {0, 0, 0, 0};
#if __clang_major__ >= 23
  i32x8 g4 = {0, 0, 0, 0, 0, 0, 0, 0};
  __builtin_amdgcn_tensor_load_to_lds(g0.v, g1.v, g2.v, g3, g4, 0);
#else
  __builtin_amdgcn_tensor_load_to_lds(g0.v, g1.v, g2.v, g3, 0);
#endif
}

// ---------------- conversion ----------------
__global__ void k_f32_to_f16(const float* __restrict__ in, _Float16* __restrict__ out, int n) {
  int i = blockIdx.x * blockDim.x + threadIdx.x;
  int stride = gridDim.x * blockDim.x;
  for (; i < n; i += stride) out[i] = (_Float16)in[i];
}

// V [B,S,D] f16 -> VT [B,H,DK,S] f16
__global__ void k_vtrans(const _Float16* __restrict__ v, _Float16* __restrict__ vt) {
  int i = blockIdx.x * blockDim.x + threadIdx.x;
  if (i >= B_ * S_ * D_) return;
  int d = i & (D_ - 1);
  int s = (i >> 9) & (S_ - 1);
  int b = i >> 18;
  int h = d >> 6, dk = d & 63;
  vt[((((size_t)b * H_ + h) * DK_ + dk) << 9) + s] = v[i];
}

// ---------------- GEMM: C[M,N] = A[M,K](f16) @ W[K,N](f16) + bias ----------------
// Block tile 128x64, 8 waves in a 4(M)x2(N) grid; each wave owns 32x32 (4 WMMAs/K-step).
// A tile (128x32) is DMA'd by the Tensor Data Mover into padded LDS (row stride 48 f16).
__global__ __launch_bounds__(256) void k_gemm(
    const _Float16* __restrict__ A, const _Float16* __restrict__ W,
    const float* __restrict__ bias,
    float* __restrict__ Cf, _Float16* __restrict__ Ch,
    int M, int N, int K, int relu) {
  __shared__ __align__(16) _Float16 As[128 * 48];
  __shared__ __align__(16) _Float16 Ws[64 * 48];
  const int t = threadIdx.x;
  const int lane = t & 31;
  const int w = t >> 5;
  const int wm = w & 3, wn = w >> 2;
  const int bm = blockIdx.y * 128, bn = blockIdx.x * 64;
  const int hf = lane >> 4;
  const int l16 = lane & 15;

  v8f c00 = {}, c01 = {}, c10 = {}, c11 = {};

  const int wk = t >> 3;        // 0..31
  const int wn0 = (t & 7) * 8;  // 0..56

  const unsigned lds_as = (unsigned)(size_t)(void*)As;

  for (int k0 = 0; k0 < K; k0 += 32) {
    if (w == 0) {
      // 128x32 f16 tile of A, padded rows: 16 DW data + 8 DW pad -> 48-f16 stride
      tdm_load_2d_f16(lds_as, A + (size_t)bm * K + k0, 32u, 128u,
                      (unsigned long long)K, (unsigned long long)K,
                      (unsigned long long)M, 3u, 7u, 1);
    }
    // W tile (32x64) loaded coalesced, stored transposed into LDS
    v8h wv = *(const v8h*)(W + (size_t)(k0 + wk) * N + bn + wn0);
#pragma unroll
    for (int j = 0; j < 8; j++) Ws[(wn0 + j) * 48 + wk] = wv[j];
    if (k0 + 32 < K) __builtin_prefetch(W + (size_t)(k0 + 32 + wk) * N + bn + wn0, 0, 1);
    if (w == 0) __builtin_amdgcn_s_wait_tensorcnt(0);
    __syncthreads();

    const _Float16* ar0 = As + (wm * 32 + l16) * 48;
    const _Float16* ar1 = ar0 + 16 * 48;
    const _Float16* br0 = Ws + (wn * 32 + l16) * 48;
    const _Float16* br1 = br0 + 16 * 48;
    v16h a0 = mk16(*(const v8h*)(ar0 + hf * 8), *(const v8h*)(ar0 + 16 + hf * 8));
    v16h a1 = mk16(*(const v8h*)(ar1 + hf * 8), *(const v8h*)(ar1 + 16 + hf * 8));
    v16h b0 = mk16(*(const v8h*)(br0 + hf * 8), *(const v8h*)(br0 + 16 + hf * 8));
    v16h b1 = mk16(*(const v8h*)(br1 + hf * 8), *(const v8h*)(br1 + 16 + hf * 8));
    c00 = __builtin_amdgcn_wmma_f32_16x16x32_f16(false, a0, false, b0, (short)0, c00, false, false);
    c01 = __builtin_amdgcn_wmma_f32_16x16x32_f16(false, a0, false, b1, (short)0, c01, false, false);
    c10 = __builtin_amdgcn_wmma_f32_16x16x32_f16(false, a1, false, b0, (short)0, c10, false, false);
    c11 = __builtin_amdgcn_wmma_f32_16x16x32_f16(false, a1, false, b1, (short)0, c11, false, false);
    __syncthreads();
  }

  const int ng0 = bn + wn * 32 + l16;
  const int ng1 = ng0 + 16;
  const float bv0 = bias ? bias[ng0] : 0.0f;
  const float bv1 = bias ? bias[ng1] : 0.0f;
#pragma unroll
  for (int r = 0; r < 8; r++) {
    int mg0 = bm + wm * 32 + r + 8 * hf;
    int mg1 = mg0 + 16;
    float v00 = c00[r] + bv0, v01 = c01[r] + bv1;
    float v10 = c10[r] + bv0, v11 = c11[r] + bv1;
    if (relu) {
      v00 = fmaxf(v00, 0.f); v01 = fmaxf(v01, 0.f);
      v10 = fmaxf(v10, 0.f); v11 = fmaxf(v11, 0.f);
    }
    if (Cf) {
      Cf[(size_t)mg0 * N + ng0] = v00; Cf[(size_t)mg0 * N + ng1] = v01;
      Cf[(size_t)mg1 * N + ng0] = v10; Cf[(size_t)mg1 * N + ng1] = v11;
    }
    if (Ch) {
      Ch[(size_t)mg0 * N + ng0] = (_Float16)v00; Ch[(size_t)mg0 * N + ng1] = (_Float16)v01;
      Ch[(size_t)mg1 * N + ng0] = (_Float16)v10; Ch[(size_t)mg1 * N + ng1] = (_Float16)v11;
    }
  }
}

// ---------------- decay attention, one block per (b,h,16-row q tile) ----------------
__global__ __launch_bounds__(256) void k_attn(
    const _Float16* __restrict__ qk,   // [B,S,D] f16 (q == k)
    const _Float16* __restrict__ vt,   // [B,H,DK,S] f16
    _Float16* __restrict__ outp,       // [B,S,D] f16
    const float* __restrict__ gammas, int layer,
    int maskDiag, int zeroPad) {
  __shared__ __align__(16) float    sc[16 * 520];
  __shared__ __align__(16) _Float16 at[16 * 528];

  const int qt = blockIdx.x;
  const int h  = blockIdx.y;
  const int b  = blockIdx.z;
  const int t = threadIdx.x, lane = t & 31, w = t >> 5;
  const int hf = lane >> 4, l16 = lane & 15;

  const float gamma = gammas[layer * H_ + h];
  const float gneg = -(gamma > 20.f ? gamma : log1pf(expf(gamma)));  // -softplus(gamma)

  const size_t baseQ = ((size_t)b * S_) * D_ + (size_t)h * DK_;

  // Phase 1: scores = Q K^T / 8
  for (int kt = w; kt < 32; kt += 8) {
    v8f acc = {};
#pragma unroll
    for (int ks = 0; ks < 2; ks++) {
      const _Float16* qp = qk + baseQ + (size_t)(qt * 16 + l16) * D_ + ks * 32 + hf * 8;
      v16h a = mk16(*(const v8h*)qp, *(const v8h*)(qp + 16));
      const _Float16* kp = qk + baseQ + (size_t)(kt * 16 + l16) * D_ + ks * 32 + hf * 8;
      v16h bb = mk16(*(const v8h*)kp, *(const v8h*)(kp + 16));
      acc = __builtin_amdgcn_wmma_f32_16x16x32_f16(false, a, false, bb, (short)0, acc, false, false);
    }
#pragma unroll
    for (int r = 0; r < 8; r++)
      sc[(r + 8 * hf) * 520 + kt * 16 + l16] = acc[r] * 0.125f;
  }
  __syncthreads();

  // Phase 2: masked softmax -> cumsum -> decay -> masked softmax
  for (int rr = w; rr < 16; rr += 8) {
    const int qg = qt * 16 + rr;
    const int limit = qg - maskDiag;  // allowed keys: key <= limit
    float* srow = sc + rr * 520;
    _Float16* arow = at + rr * 528;
    const bool rowzero = (limit < 0) || (zeroPad && qg == 0);

    float sv[16]; bool val[16];
    float mx = -3.0e38f;
#pragma unroll
    for (int j = 0; j < 16; j++) {
      int key = lane * 16 + j;
      sv[j] = srow[key];
      val[j] = (key <= limit);
      if (val[j]) mx = fmaxf(mx, sv[j]);
    }
    mx = wave_max(mx);

    float p[16]; float lsum = 0.f;
#pragma unroll
    for (int j = 0; j < 16; j++) { p[j] = val[j] ? expf(sv[j] - mx) : 0.f; lsum += p[j]; }
    float tot = wave_sum(lsum);
    float inv = 1.0f / fmaxf(tot, 1e-30f);

    float c[16]; float run = 0.f;
#pragma unroll
    for (int j = 0; j < 16; j++) { run += p[j]; c[j] = run; }
    float incl = wave_incl_scan(run);
    float offs = incl - run;

    float s2[16]; float mx2 = -3.0e38f;
#pragma unroll
    for (int j = 0; j < 16; j++) {
      int key = lane * 16 + j;
      float cumn = (offs + c[j]) * inv;  // normalized cumsum(p)
      float rem = 1.0f - cumn;           // disttot - distcum
      float pos = fabsf((float)(qg - key));
      float dist = sqrtf(fmaxf(rem * pos, 0.f));
      float eff = expf(dist * gneg);
      eff = fminf(fmaxf(eff, 1e-5f), 1e5f);
      if (val[j]) { s2[j] = sv[j] * eff; mx2 = fmaxf(mx2, s2[j]); } else s2[j] = 0.f;
    }
    mx2 = wave_max(mx2);

    float e2[16]; float lsum2 = 0.f;
#pragma unroll
    for (int j = 0; j < 16; j++) { e2[j] = val[j] ? expf(s2[j] - mx2) : 0.f; lsum2 += e2[j]; }
    float tot2 = wave_sum(lsum2);
    float inv2 = 1.0f / fmaxf(tot2, 1e-30f);
#pragma unroll
    for (int j = 0; j < 16; j++) {
      float a = rowzero ? 0.f : e2[j] * inv2;
      arow[lane * 16 + j] = (_Float16)a;
    }
  }
  __syncthreads();

  // Phase 3: out = attn @ V  (waves 0..3 each own one 16-col dk tile)
  if (w < 4) {
    const int nt = w;
    v8f acc = {};
    for (int ks = 0; ks < 16; ks++) {
      const _Float16* ap = at + l16 * 528 + ks * 32 + hf * 8;
      v16h a = mk16(*(const v8h*)ap, *(const v8h*)(ap + 16));
      const _Float16* bp = vt + ((((size_t)b * H_ + h) * DK_) + nt * 16 + l16) * S_ + ks * 32 + hf * 8;
      v16h bb = mk16(*(const v8h*)bp, *(const v8h*)(bp + 16));
      acc = __builtin_amdgcn_wmma_f32_16x16x32_f16(false, a, false, bb, (short)0, acc, false, false);
    }
#pragma unroll
    for (int r = 0; r < 8; r++) {
      int m = r + 8 * hf;
      int sg = qt * 16 + m;
      int d = h * DK_ + nt * 16 + l16;
      outp[((size_t)b * S_ + sg) * D_ + d] = (_Float16)acc[r];
    }
  }
}

// ---------------- LayerNorm: out = LN(res + proj) * g + b ----------------
__global__ __launch_bounds__(256) void k_ln(
    const float* __restrict__ res, const float* __restrict__ proj,
    const float* __restrict__ g, const float* __restrict__ bta,
    float* __restrict__ out32, _Float16* __restrict__ out16) {
  const int w = threadIdx.x >> 5, lane = threadIdx.x & 31;
  const int row = blockIdx.x * 8 + w;
  const float* r0 = res + (size_t)row * D_;
  const float* p0 = proj + (size_t)row * D_;
  float x[16]; float s = 0.f;
#pragma unroll
  for (int j = 0; j < 16; j++) { x[j] = r0[lane * 16 + j] + p0[lane * 16 + j]; s += x[j]; }
  s = wave_sum(s);
  float mean = s * (1.0f / D_);
  float v = 0.f;
#pragma unroll
  for (int j = 0; j < 16; j++) { float d = x[j] - mean; v += d * d; }
  v = wave_sum(v) * (1.0f / D_);
  float inv = rsqrtf(v + 1e-5f);
#pragma unroll
  for (int j = 0; j < 16; j++) {
    int d = lane * 16 + j;
    float y = (x[j] - mean) * inv * g[d] + bta[d];
    if (out32) out32[(size_t)row * D_ + d] = y;
    if (out16) out16[(size_t)row * D_ + d] = (_Float16)y;
  }
}

extern "C" void kernel_launch(void* const* d_in, const int* in_sizes, int n_in,
                              void* d_out, int out_size, void* d_ws, size_t ws_size,
                              hipStream_t stream) {
  (void)in_sizes; (void)n_in; (void)out_size; (void)ws_size;
  const float* q_embed  = (const float*)d_in[0];
  const float* qa_embed = (const float*)d_in[1];
  const float* Wk = (const float*)d_in[2];
  const float* bk = (const float*)d_in[3];
  const float* Wv = (const float*)d_in[4];
  const float* bv = (const float*)d_in[5];
  const float* Wo = (const float*)d_in[6];
  const float* bo = (const float*)d_in[7];
  const float* gammas = (const float*)d_in[8];
  const float* W1 = (const float*)d_in[9];
  const float* b1 = (const float*)d_in[10];
  const float* W2 = (const float*)d_in[11];
  const float* b2 = (const float*)d_in[12];
  const float* ln1_g = (const float*)d_in[13];
  const float* ln1_b = (const float*)d_in[14];
  const float* ln2_g = (const float*)d_in[15];
  const float* ln2_b = (const float*)d_in[16];
  float* out = (float*)d_out;

  char* ws = (char*)d_ws;
  size_t off = 0;
  auto alloc = [&](size_t bytes) -> char* {
    char* p = ws + off;
    off += (bytes + 255) & ~(size_t)255;
    return p;
  };

  const int MD = B_ * S_;              // 8192 rows
  const size_t NACT = (size_t)MD * D_; // 4M elements

  _Float16* Wk16 = (_Float16*)alloc(3ull * D_ * D_ * 2);
  _Float16* Wv16 = (_Float16*)alloc(3ull * D_ * D_ * 2);
  _Float16* Wo16 = (_Float16*)alloc(3ull * D_ * D_ * 2);
  _Float16* W116 = (_Float16*)alloc(3ull * D_ * FF_ * 2);
  _Float16* W216 = (_Float16*)alloc(3ull * FF_ * D_ * 2);
  _Float16* qa16 = (_Float16*)alloc(NACT * 2);
  _Float16* q16  = (_Float16*)alloc(NACT * 2);
  _Float16* qkh  = (_Float16*)alloc(NACT * 2);
  _Float16* vh   = (_Float16*)alloc(NACT * 2);
  _Float16* vth  = (_Float16*)alloc(NACT * 2);
  _Float16* aoh  = (_Float16*)alloc(NACT * 2);
  _Float16* t16  = (_Float16*)alloc(NACT * 2);
  _Float16* y16  = (_Float16*)alloc(NACT * 2);
  _Float16* x16  = (_Float16*)alloc(NACT * 2);
  _Float16* ff1h = (_Float16*)alloc((size_t)MD * FF_ * 2);
  float* proj32  = (float*)alloc(NACT * 4);
  float* t32     = (float*)alloc(NACT * 4);
  float* x32     = (float*)alloc(NACT * 4);
  float* ff232   = (float*)alloc(NACT * 4);
  float* dump32  = (float*)alloc(NACT * 4);

  auto cvt = [&](const float* src, _Float16* dst, int n) {
    k_f32_to_f16<<<dim3((n + 255) / 256), dim3(256), 0, stream>>>(src, dst, n);
  };
  auto gemm = [&](const _Float16* A, const _Float16* Wt, const float* bias,
                  float* Cf, _Float16* Ch, int M, int N, int K, int relu) {
    k_gemm<<<dim3(N / 64, M / 128), dim3(256), 0, stream>>>(A, Wt, bias, Cf, Ch, M, N, K, relu);
  };
  auto vtrans = [&](const _Float16* src, _Float16* dst) {
    k_vtrans<<<dim3((B_ * S_ * D_ + 255) / 256), dim3(256), 0, stream>>>(src, dst);
  };
  auto attn = [&](const _Float16* qkp, const _Float16* vtp, _Float16* op,
                  int layer, int maskDiag, int zp) {
    k_attn<<<dim3(S_ / 16, H_, B_), dim3(256), 0, stream>>>(qkp, vtp, op, gammas, layer, maskDiag, zp);
  };
  auto ln = [&](const float* res, const float* proj, const float* g, const float* bb,
                float* o32, _Float16* o16) {
    k_ln<<<dim3(MD / 8), dim3(256), 0, stream>>>(res, proj, g, bb, o32, o16);
  };

  // conversions
  cvt(qa_embed, qa16, (int)NACT);
  cvt(q_embed,  q16,  (int)NACT);
  cvt(Wk, Wk16, 3 * D_ * D_);
  cvt(Wv, Wv16, 3 * D_ * D_);
  cvt(Wo, Wo16, 3 * D_ * D_);
  cvt(W1, W116, 3 * D_ * FF_);
  cvt(W2, W216, 3 * FF_ * D_);

  const int WDD = D_ * D_;

  // ---- Layer 0: tl(0, qa, qa, qa, mask1, zero_pad=false, apply_pos=true) -> y
  gemm(qa16, Wk16, bk, nullptr, qkh, MD, D_, D_, 0);
  gemm(qa16, Wv16, bv, nullptr, vh,  MD, D_, D_, 0);
  vtrans(vh, vth);
  attn(qkh, vth, aoh, 0, /*maskDiag=*/0, /*zp=*/0);
  gemm(aoh, Wo16, bo, proj32, nullptr, MD, D_, D_, 0);
  ln(qa_embed, proj32, ln1_g, ln1_b, t32, t16);
  gemm(t16, W116, b1, nullptr, ff1h, MD, FF_, D_, 1);
  gemm(ff1h, W216, b2, ff232, nullptr, MD, D_, FF_, 0);
  ln(t32, ff232, ln2_g, ln2_b, dump32, y16);

  // ---- Layer 1: tl(1, q, q, q, mask1, zero_pad=false, apply_pos=false) -> x
  gemm(q16, Wk16 + WDD, bk + D_, nullptr, qkh, MD, D_, D_, 0);
  gemm(q16, Wv16 + WDD, bv + D_, nullptr, vh,  MD, D_, D_, 0);
  vtrans(vh, vth);
  attn(qkh, vth, aoh, 1, /*maskDiag=*/0, /*zp=*/0);
  gemm(aoh, Wo16 + WDD, bo + D_, proj32, nullptr, MD, D_, D_, 0);
  ln(q_embed, proj32, ln1_g + D_, ln1_b + D_, x32, x16);

  // ---- Layer 2: tl(2, x, x, y, mask0, zero_pad=true, apply_pos=true) -> out
  gemm(x16, Wk16 + 2 * WDD, bk + 2 * D_, nullptr, qkh, MD, D_, D_, 0);
  gemm(y16, Wv16 + 2 * WDD, bv + 2 * D_, nullptr, vh,  MD, D_, D_, 0);
  vtrans(vh, vth);
  attn(qkh, vth, aoh, 2, /*maskDiag=*/1, /*zp=*/1);
  gemm(aoh, Wo16 + 2 * WDD, bo + 2 * D_, proj32, nullptr, MD, D_, D_, 0);
  ln(x32, proj32, ln1_g + 2 * D_, ln1_b + 2 * D_, t32, t16);
  gemm(t16, W116 + 2 * D_ * FF_, b1 + 2 * FF_, nullptr, ff1h, MD, FF_, D_, 1);
  gemm(ff1h, W216 + 2 * FF_ * D_, b2 + 2 * D_, ff232, nullptr, MD, D_, FF_, 0);
  ln(t32, ff232, ln2_g + 2 * D_, ln2_b + 2 * D_, out, t16);
}
